// Static_STGAT_40029095199106
// MI455X (gfx1250) — compile-verified, matrix operand order checked by reading, not run on hardware
//
#include <hip/hip_runtime.h>
#include <hip/hip_bf16.h>
#include <math.h>

// ---------------- types ----------------
typedef __bf16 bf16;
typedef __attribute__((ext_vector_type(16))) __bf16 v16bf;
typedef __attribute__((ext_vector_type(8)))  float  v8f;

// problem constants
#define B_   8
#define T_   64
#define N_   256
#define F_   64
#define C_   128
#define O_   32
#define NC_  16
#define H_   1024
#define NB_  32
#define G_   8
#define BTN_ (B_*T_*N_)      // 131072
#define TN_  (T_*N_)         // 16384

// ---------------- CDNA5 async global->LDS copy ----------------
// GLOBAL_LOAD_ASYNC_TO_LDS_B128: per-lane 16B DMA into LDS, tracked by ASYNCcnt.
__device__ __forceinline__ void async_copy_b128(void* lds_dst, const void* gsrc) {
  unsigned lds_off = (unsigned)(size_t)lds_dst;   // low 32 bits of flat LDS addr = LDS offset
  asm volatile("global_load_async_to_lds_b128 %0, %1, off"
               :: "v"(lds_off), "v"(gsrc) : "memory");
}
__device__ __forceinline__ void wait_async_all() {
  asm volatile("s_wait_asynccnt 0x0" ::: "memory");
}

// ---------------- f32 -> bf16 convert ----------------
__global__ __launch_bounds__(256)
void k_f32_to_bf16(const float* __restrict__ in, bf16* __restrict__ out, size_t n) {
  size_t i = (size_t)blockIdx.x * blockDim.x + threadIdx.x;
  size_t stride = (size_t)gridDim.x * blockDim.x;
  for (; i < n; i += stride) out[i] = (bf16)in[i];
}

__global__ __launch_bounds__(256)
void k_zero_u32(unsigned* __restrict__ p, int n) {
  int i = blockIdx.x * blockDim.x + threadIdx.x;
  if (i < n) p[i] = 0u;
}

// ---------------- bf16 WMMA GEMM, async double-buffered ----------------
// D[m][n] = sum_k A[m][k]*W[n][k] + bias[n] (+ C0[m*ldc0+n])
// A: M x K row-major bf16; W: N x K row-major bf16. K % 64 == 0.
// OOB rows are CLAMPED on load (branch-free); stores guarded to [0,Ms) x [0,N).
// Block = 256 threads (8 waves). Block tile: 32 x BN. Wave tile: 16 x (BN/4).
// grid: x = ceil(N/BN), y = ceil(M/32)
template<int BN>
__global__ __launch_bounds__(256)
void k_gemm_bf16(const bf16* __restrict__ A, const bf16* __restrict__ W,
                 const float* __restrict__ bias,
                 const float* __restrict__ C0, long ldc0,
                 float* __restrict__ outF, bf16* __restrict__ outB,
                 int M, int Ms, int N, int K)
{
  constexpr int KB = 64;                 // K per stage
  constexpr int NT = BN / 64;            // 16-col accumulator tiles per wave
  constexpr int LDSTR = KB + 8;          // 72 elems = 144B row stride (9x16B)
  __shared__ __align__(16) bf16 sA[2][32][LDSTR];
  __shared__ __align__(16) bf16 sW[2][BN][LDSTR];

  const int tid  = threadIdx.x;
  const int wave = tid >> 5;
  const int lane = tid & 31;
  const int m0 = blockIdx.y * 32;
  const int n0 = blockIdx.x * BN;
  const int wm = (wave & 1) * 16;        // wave m-tile
  const int wn = (wave >> 1) * (BN / 4); // wave n-strip
  const int hl = lane >> 4;
  const int mr = lane & 15;

  v8f zero = {};
  v8f acc[NT];
#pragma unroll
  for (int i = 0; i < NT; ++i) acc[i] = zero;

  const int nk = K / KB;

  // branch-free staging of one 32xKB A tile and BN x KB W tile via async DMA
  auto stage = [&](int buf, int ks) {
    const size_t kofs = (size_t)ks * KB;
    {   // A: 2048 elems, one b128 per thread
      int e = tid * 8;
      int r = e >> 6, c = e & 63;
      int gm = m0 + r; gm = (gm < M) ? gm : (M - 1);         // clamp
      async_copy_b128(&sA[buf][r][c], A + (size_t)gm * K + kofs + c);
    }
#pragma unroll
    for (int q = 0; q < BN / 32; ++q) {  // W: BN*64 elems, BN/32 b128 per thread
      int e = tid * (BN / 4) + q * 8;
      int r = e >> 6, c = e & 63;
      int gn = n0 + r; gn = (gn < N) ? gn : (N - 1);         // clamp
      async_copy_b128(&sW[buf][r][c], W + (size_t)gn * K + kofs + c);
    }
  };

  stage(0, 0);
  for (int ks = 0; ks < nk; ++ks) {
    const int cur = ks & 1;
    wait_async_all();        // our async stores into buffer `cur` are done
    __syncthreads();         // everyone's are
    if (ks + 1 < nk) stage(cur ^ 1, ks + 1);        // overlap next copy with math
    if (ks + 2 < nk) {                              // keep L2 streaming (prefetch_b8)
      __builtin_prefetch(W + (size_t)(n0 + (tid & (BN - 1))) * K + (size_t)(ks + 2) * KB, 0, 1);
    }

#pragma unroll
    for (int h2 = 0; h2 < 2; ++h2) {     // two k-halves of 32
      v16bf a;
#pragma unroll
      for (int v = 0; v < 8; ++v) {      // ISA 16-bit A layout
        int kk = ((v < 4) ? (2 * v) : (16 + 2 * (v - 4))) + 8 * hl + 32 * h2;
        a[2 * v]     = sA[cur][wm + mr][kk];
        a[2 * v + 1] = sA[cur][wm + mr][kk + 1];
      }
#pragma unroll
      for (int nt = 0; nt < NT; ++nt) {
        v16bf b;
#pragma unroll
        for (int v = 0; v < 8; ++v) {
          int kk = ((v < 4) ? (2 * v) : (16 + 2 * (v - 4))) + 8 * hl + 32 * h2;
          b[2 * v]     = sW[cur][wn + nt * 16 + mr][kk];
          b[2 * v + 1] = sW[cur][wn + nt * 16 + mr][kk + 1];
        }
        acc[nt] = __builtin_amdgcn_wmma_f32_16x16x32_bf16(false, a, false, b,
                                                          (short)0, acc[nt], false, false);
      }
    }
  }

  // store: lanes 0-15 -> M=r, N=lane; lanes 16-31 -> M=8+r, N=lane-16
#pragma unroll
  for (int nt = 0; nt < NT; ++nt) {
#pragma unroll
    for (int r = 0; r < 8; ++r) {
      int mm = m0 + wm + r + 8 * hl;
      int nn = n0 + wn + nt * 16 + mr;
      if (mm < Ms && nn < N) {
        float val = acc[nt][r];
        if (bias) val += bias[nn];
        if (C0)   val += C0[(size_t)mm * ldc0 + nn];
        if (outF) outF[(size_t)mm * N + nn] = val;
        if (outB) outB[(size_t)mm * N + nn] = (bf16)val;
      }
    }
  }
}

// ---------------- GATv2 attention per (t, g) block ----------------
__device__ __forceinline__ float sigmoidf_(float x) { return 1.0f / (1.0f + expf(-x)); }

__global__ __launch_bounds__(128)
void k_gat_attn(const float* __restrict__ gl0,   // (T,N,C)
                const float* __restrict__ gr0,   // (T,N,C)
                const float* __restrict__ V,     // (N,N)
                const float* __restrict__ we, const float* __restrict__ att,
                float* __restrict__ out0)        // (T,N,C)
{
  const int t = blockIdx.x;
  const int g = blockIdx.y;
  __shared__ float sGl[NB_][C_ + 1];
  __shared__ float sGr[NB_][C_ + 1];
  __shared__ float sAeff[NB_][NB_];   // [j][i]
  __shared__ float sS[NB_][NB_];      // [i][j]
  __shared__ float sWe[C_], sAtt[C_];

  const int tid = threadIdx.x;  // 128
  const int rowbase = t * N_ + g * NB_;

  sWe[tid]  = we[tid];
  sAtt[tid] = att[tid];
  for (int r = 0; r < NB_; ++r) {
    sGl[r][tid] = gl0[(size_t)(rowbase + r) * C_ + tid];
    sGr[r][tid] = gr0[(size_t)(rowbase + r) * C_ + tid];
  }
#pragma unroll
  for (int k = 0; k < 8; ++k) {
    int e = tid + 128 * k;        // 0..1023
    int j = e >> 5, i = e & 31;
    sAeff[j][i] = sigmoidf_(V[(size_t)(g * NB_ + j) * N_ + (g * NB_ + i)]);
  }
  __syncthreads();

  float selfv = 0.0f;
  if (tid < NB_) {
    float s = 0.0f;
    for (int j = 0; j < NB_; ++j) s += sAeff[j][tid];
    selfv = (s - sAeff[tid][tid]) * (1.0f / (NB_ - 1));
  }
  __syncthreads();
  if (tid < NB_) sAeff[tid][tid] = selfv;
  __syncthreads();

#pragma unroll
  for (int k = 0; k < 8; ++k) {
    int e = tid + 128 * k;
    int i = e >> 5, j = e & 31;
    float aeff = sAeff[j][i];
    float s = 0.0f;
    for (int c = 0; c < C_; ++c) {
      float pre = sGl[j][c] + sGr[i][c] + aeff * sWe[c];
      float lr = (pre > 0.0f) ? pre : 0.2f * pre;
      s += lr * sAtt[c];
    }
    sS[i][j] = s;
  }
  __syncthreads();

  if (tid < NB_) {
    int i = tid;
    float mx = sS[i][0];
    for (int j = 1; j < NB_; ++j) mx = fmaxf(mx, sS[i][j]);
    float sum = 0.0f;
    for (int j = 0; j < NB_; ++j) { float e = expf(sS[i][j] - mx); sS[i][j] = e; sum += e; }
    float inv = 1.0f / sum;
    for (int j = 0; j < NB_; ++j) sS[i][j] *= inv;
  }
  __syncthreads();

  int c = tid;
  for (int i = 0; i < NB_; ++i) {
    float acc = 0.0f;
    for (int j = 0; j < NB_; ++j) acc += sS[i][j] * sGl[j][c];
    out0[(size_t)(rowbase + i) * C_ + c] = acc;
  }
}

// ---------------- assemble GAT output (+gat_bias) into bf16 ----------------
__global__ __launch_bounds__(256)
void k_assemble_out(const float* __restrict__ gl,    // (B,T,N,C)
                    const float* __restrict__ out0,  // (T,N,C) (== b=0 block)
                    const float* __restrict__ gat_bias,
                    bf16* __restrict__ out_bf, size_t n)
{
  size_t i = (size_t)blockIdx.x * blockDim.x + threadIdx.x;
  size_t stride = (size_t)gridDim.x * blockDim.x;
  const size_t b0n = (size_t)TN_ * C_;
  for (; i < n; i += stride) {
    int c = (int)(i & (C_ - 1));
    float v = (i < b0n ? out0[i] : gl[i]) + gat_bias[c];
    out_bf[i] = (bf16)v;
  }
}

// ---------------- LSTM cell (gates -> c,h) ----------------
__global__ __launch_bounds__(256)
void k_lstm_cell(const float* __restrict__ g,   // (8,4H) i|f|g|o
                 float* __restrict__ c,         // (8,H)
                 bf16* __restrict__ h_bf,       // (16,H) rows 8..15 stay zero
                 bf16* __restrict__ h_seq_bf,   // (B,T,H) or null
                 float* __restrict__ h_last,    // (8,H) or null
                 int t)
{
  int idx = blockIdx.x * blockDim.x + threadIdx.x;
  if (idx >= B_ * H_) return;
  int b = idx >> 10, n = idx & (H_ - 1);
  const float* gr = g + (size_t)b * (4 * H_);
  float ig = sigmoidf_(gr[n]);
  float fg = sigmoidf_(gr[n + H_]);
  float gg = tanhf(gr[n + 2 * H_]);
  float og = sigmoidf_(gr[n + 3 * H_]);
  float cv = fg * c[idx] + ig * gg;
  c[idx] = cv;
  float hv = og * tanhf(cv);
  h_bf[(size_t)b * H_ + n] = (bf16)hv;
  if (h_seq_bf) h_seq_bf[((size_t)b * T_ + t) * H_ + n] = (bf16)hv;
  if (h_last)   h_last[idx] = hv;
}

// ---------------- final classifier (8x16, K=1024) ----------------
__global__ __launch_bounds__(128)
void k_final(const float* __restrict__ h_last, const float* __restrict__ Wout,
             const float* __restrict__ bout, float* __restrict__ out)
{
  int tid = threadIdx.x;          // 128 = 8*16
  int b = tid >> 4, nc = tid & 15;
  float acc = bout[nc];
  const float* h = h_last + (size_t)b * H_;
  const float* w = Wout + (size_t)nc * H_;
  for (int k = 0; k < H_; ++k) acc += h[k] * w[k];
  out[b * NC_ + nc] = acc;
}

// ---------------- host launcher ----------------
static inline size_t align256(size_t x) { return (x + 255) & ~(size_t)255; }

extern "C" void kernel_launch(void* const* d_in, const int* in_sizes, int n_in,
                              void* d_out, int out_size, void* d_ws, size_t ws_size,
                              hipStream_t stream) {
  (void)in_sizes; (void)n_in; (void)out_size; (void)ws_size;
  const float* X        = (const float*)d_in[0];
  const float* V_Adap   = (const float*)d_in[1];
  const float* Wl       = (const float*)d_in[2];
  const float* bl       = (const float*)d_in[3];
  const float* Wr       = (const float*)d_in[4];
  const float* br       = (const float*)d_in[5];
  const float* att      = (const float*)d_in[6];
  const float* we       = (const float*)d_in[7];
  const float* gat_bias = (const float*)d_in[8];
  const float* Wfc      = (const float*)d_in[9];
  const float* bfc      = (const float*)d_in[10];
  const float* Wih0     = (const float*)d_in[11];
  const float* Whh0     = (const float*)d_in[12];
  const float* bih0     = (const float*)d_in[13];
  const float* bhh0     = (const float*)d_in[14];
  const float* Wih1     = (const float*)d_in[15];
  const float* Whh1     = (const float*)d_in[16];
  const float* bih1     = (const float*)d_in[17];
  const float* bhh1     = (const float*)d_in[18];
  const float* Wout     = (const float*)d_in[19];
  const float* bout     = (const float*)d_in[20];

  // -------- workspace layout --------
  char* ws = (char*)d_ws;
  size_t off = 0;
  auto take = [&](size_t bytes) { char* p = ws + off; off = align256(off + bytes); return p; };

  bf16* X_bf    = (bf16*)take((size_t)BTN_ * F_ * 2);
  bf16* Wl_bf   = (bf16*)take((size_t)C_ * F_ * 2);
  bf16* Wr_bf   = (bf16*)take((size_t)C_ * F_ * 2);
  bf16* Wfc_bf  = (bf16*)take((size_t)O_ * C_ * 2);
  bf16* Wih0_bf = (bf16*)take((size_t)4 * H_ * (N_ * O_) * 2);
  bf16* Whh0_bf = (bf16*)take((size_t)4 * H_ * H_ * 2);
  bf16* Wih1_bf = (bf16*)take((size_t)4 * H_ * H_ * 2);
  bf16* Whh1_bf = (bf16*)take((size_t)4 * H_ * H_ * 2);
  float* gl     = (float*)take((size_t)BTN_ * C_ * 4);
  float* gr0    = (float*)take((size_t)TN_ * C_ * 4);
  float* out0   = (float*)take((size_t)TN_ * C_ * 4);
  bf16* out_bf  = (bf16*)take((size_t)BTN_ * C_ * 2);
  bf16* s_bf    = (bf16*)take((size_t)B_ * T_ * (N_ * O_) * 2);
  float* xg     = (float*)take((size_t)B_ * T_ * 4 * H_ * 4);
  float* gbuf   = (float*)take((size_t)B_ * 4 * H_ * 4);
  float* cbuf   = (float*)take((size_t)B_ * H_ * 4);
  bf16* hbf     = (bf16*)take((size_t)16 * H_ * 2);
  bf16* h1seq   = (bf16*)take((size_t)B_ * T_ * H_ * 2);
  float* hlast  = (float*)take((size_t)B_ * H_ * 4);

  const int CT = 256;
  auto cvt = [&](const float* src, bf16* dst, size_t n) {
    int blocks = (int)((n + CT - 1) / CT); if (blocks > 32768) blocks = 32768;
    k_f32_to_bf16<<<blocks, CT, 0, stream>>>(src, dst, n);
  };

  // -------- bf16 conversions --------
  cvt(X,    X_bf,    (size_t)BTN_ * F_);
  cvt(Wl,   Wl_bf,   (size_t)C_ * F_);
  cvt(Wr,   Wr_bf,   (size_t)C_ * F_);
  cvt(Wfc,  Wfc_bf,  (size_t)O_ * C_);
  cvt(Wih0, Wih0_bf, (size_t)4 * H_ * (N_ * O_));
  cvt(Whh0, Whh0_bf, (size_t)4 * H_ * H_);
  cvt(Wih1, Wih1_bf, (size_t)4 * H_ * H_);
  cvt(Whh1, Whh1_bf, (size_t)4 * H_ * H_);

  auto gemmS = [&](const bf16* A, const bf16* W, const float* bias,
                   const float* C0, long ldc0, float* oF, bf16* oB,
                   int M, int Ms, int N, int K) {
    dim3 grid((N + 63) / 64, (M + 31) / 32);
    k_gemm_bf16<64><<<grid, 256, 0, stream>>>(A, W, bias, C0, ldc0, oF, oB, M, Ms, N, K);
  };
  auto gemmB = [&](const bf16* A, const bf16* W, const float* bias,
                   float* oF, bf16* oB, int M, int N, int K) {
    dim3 grid((N + 255) / 256, (M + 31) / 32);
    k_gemm_bf16<256><<<grid, 256, 0, stream>>>(A, W, bias, nullptr, 0, oF, oB, M, M, N, K);
  };

  // -------- gl (all samples), gr (sample 0 only) --------
  gemmS(X_bf, Wl_bf, bl, nullptr, 0, gl,  nullptr, BTN_, BTN_, C_, F_);
  gemmS(X_bf, Wr_bf, br, nullptr, 0, gr0, nullptr, TN_,  TN_,  C_, F_);

  // -------- GATv2 attention on diagonal blocks (sample 0) --------
  k_gat_attn<<<dim3(T_, G_), 128, 0, stream>>>(gl, gr0, V_Adap, we, att, out0);

  // -------- assemble (out0 | gl) + gat_bias -> bf16 --------
  {
    size_t n = (size_t)BTN_ * C_;
    int blocks = (int)((n + CT - 1) / CT); if (blocks > 32768) blocks = 32768;
    k_assemble_out<<<blocks, CT, 0, stream>>>(gl, out0, gat_bias, out_bf, n);
  }

  // -------- per-node FC -> s (bf16, (B,T,8192)) --------
  gemmS(out_bf, Wfc_bf, bfc, nullptr, 0, nullptr, s_bf, BTN_, BTN_, O_, C_);

  // -------- LSTM0 input GEMM (dominant, 34 GFLOP) --------
  gemmB(s_bf, Wih0_bf, bih0, xg, nullptr, B_ * T_, 4 * H_, N_ * O_);

  // -------- LSTM0 recurrence --------
  k_zero_u32<<<(B_ * H_ + 255) / 256, 256, 0, stream>>>((unsigned*)cbuf, B_ * H_);
  k_zero_u32<<<(16 * H_ / 2 + 255) / 256, 256, 0, stream>>>((unsigned*)hbf, 16 * H_ / 2);
  for (int t = 0; t < T_; ++t) {
    gemmS(hbf, Whh0_bf, bhh0, xg + (size_t)t * 4 * H_, (long)T_ * 4 * H_,
          gbuf, nullptr, 16, B_, 4 * H_, H_);
    k_lstm_cell<<<(B_ * H_ + 255) / 256, 256, 0, stream>>>(gbuf, cbuf, hbf, h1seq, nullptr, t);
  }

  // -------- LSTM1 input GEMM --------
  gemmB(h1seq, Wih1_bf, bih1, xg, nullptr, B_ * T_, 4 * H_, H_);

  // -------- LSTM1 recurrence --------
  k_zero_u32<<<(B_ * H_ + 255) / 256, 256, 0, stream>>>((unsigned*)cbuf, B_ * H_);
  k_zero_u32<<<(16 * H_ / 2 + 255) / 256, 256, 0, stream>>>((unsigned*)hbf, 16 * H_ / 2);
  for (int t = 0; t < T_; ++t) {
    gemmS(hbf, Whh1_bf, bhh1, xg + (size_t)t * 4 * H_, (long)T_ * 4 * H_,
          gbuf, nullptr, 16, B_, 4 * H_, H_);
    k_lstm_cell<<<(B_ * H_ + 255) / 256, 256, 0, stream>>>(gbuf, cbuf, hbf, nullptr,
                                                           (t == T_ - 1) ? hlast : nullptr, t);
  }

  // -------- classifier --------
  k_final<<<1, 128, 0, stream>>>(hlast, Wout, bout, (float*)d_out);
}
